// Net_24180665876499
// MI455X (gfx1250) — matrix operand order repaired and checked by reading, not compiled, not run on hardware
//
#include <hip/hip_runtime.h>
#include <hip/hip_bf16.h>
#include <cstdint>

typedef __attribute__((ext_vector_type(16))) _Float16 v16h;
typedef __attribute__((ext_vector_type(8)))  _Float16 v8h;
typedef __attribute__((ext_vector_type(8)))  float    v8f;

#define EW_GRID 2048
#define EW_BLK  256
#define RED_NB  1024

// ---------------------------------------------------------------------------
// LDS-staged WMMA GEMM: C[m,n] = sum_k A[m,k] * W[n,k] + bias[n]
//  - block = 256 threads (8 waves), block tile = 128 rows x 64 cols
//  - the 64 x Kpad weight slab is staged into LDS once per block (<=45KB)
//  - each wave owns one 16-row m-tile and 4 accumulators (4 n-tiles), so one
//    A fragment feeds 4 v_wmma_f32_16x16x32_f16 issues per k-step
//  - weight buffers are zero-padded to 64-row multiples: staging never goes
//    OOB and EXEC stays all-ones through every WMMA (ISA requirement);
//    only the f32 stores are guarded by n < N
// ---------------------------------------------------------------------------
__launch_bounds__(256)
__global__ void wmma_gemm_lds(const _Float16* __restrict__ A, int lda,
                              const _Float16* __restrict__ W, int ldw,
                              const float* __restrict__ bias,
                              float* __restrict__ C, int ldc,
                              int nbn, long M, int N, int Kpad) {
  extern __shared__ _Float16 Bs[];           // 64 * Kpad halves
  const int tid = threadIdx.x;
  const long bx = blockIdx.x % nbn;          // n block
  const long by = blockIdx.x / nbn;          // m block
  const int  tn0 = (int)bx * 64;
  const long tm0 = by * 128;

  // ---- cooperative stage of B block (64 output features x Kpad) ----
  const int kvec = Kpad >> 3;                // 16B chunks per weight row
  for (int cid = tid; cid < 64 * kvec; cid += 256) {
    const int n  = cid / kvec;
    const int ko = (cid - n * kvec) << 3;
    *(v8h*)(Bs + n * Kpad + ko) = *(const v8h*)(W + (long)(tn0 + n) * ldw + ko);
  }
  __syncthreads();

  const int wave = tid >> 5;
  const int lane = tid & 31;
  const int r = lane & 15;
  const int g = lane >> 4;
  const _Float16* __restrict__ arow = A + (tm0 + wave * 16 + r) * (long)lda;

  v8f acc[4] = {};
  for (int k0 = 0; k0 < Kpad; k0 += 32) {
    // A 16x32 f16 fragment: lane (r,g) holds K = k0+8g+0..7 and k0+16+8g+0..7
    v16h a;
    {
      v8h lo = *(const v8h*)(arow + k0 + 8 * g);
      v8h hi = *(const v8h*)(arow + k0 + 16 + 8 * g);
#pragma unroll
      for (int e = 0; e < 8; ++e) { a[e] = lo[e]; a[8 + e] = hi[e]; }
    }
#pragma unroll
    for (int j = 0; j < 4; ++j) {
      // B 32x16 fragment: lane holds col n = 16j + r, K = k0+8g+0..7 and k0+16+8g+0..7 (LDS)
      const _Float16* bp = Bs + (j * 16 + r) * Kpad + k0;
      v8h blo = *(const v8h*)(bp + 8 * g);
      v8h bhi = *(const v8h*)(bp + 16 + 8 * g);
      v16h b;
#pragma unroll
      for (int e = 0; e < 8; ++e) { b[e] = blo[e]; b[8 + e] = bhi[e]; }
      acc[j] = __builtin_amdgcn_wmma_f32_16x16x32_f16(false, a, false, b,
                                                      (short)0, acc[j], false, false);
    }
  }

#pragma unroll
  for (int j = 0; j < 4; ++j) {
    const int cn = tn0 + j * 16 + r;         // C/D: lane -> column n
    if (cn < N) {
      const float bv = bias[cn];
#pragma unroll
      for (int jj = 0; jj < 8; ++jj) {       // VGPR jj -> row m = jj + 8*g
        const long m = tm0 + wave * 16 + 8 * g + jj;
        C[m * (long)ldc + cn] = acc[j][jj] + bv;
      }
    }
  }
}

// ---------------------------------------------------------------------------
// Weight / input conversion (weights padded to Npad64 rows x Kpad cols)
// ---------------------------------------------------------------------------
__global__ void cvt_w(const float* __restrict__ Wf, int N, int K,
                      _Float16* __restrict__ Wh, int Npad, int Kpad) {
  const long total = (long)Npad * Kpad;
  const long stride = (long)gridDim.x * blockDim.x;
  for (long i = (long)blockIdx.x * blockDim.x + threadIdx.x; i < total; i += stride) {
    const int n = (int)(i / Kpad), k = (int)(i % Kpad);
    Wh[i] = (n < N && k < K) ? (_Float16)Wf[(long)n * K + k] : (_Float16)0.f;
  }
}

__global__ void cvt_s1(const float* __restrict__ state, _Float16* __restrict__ out) {
  // s1[b,d,c] = state[((b*2+0)*256 + d)*128 + c]; out (65536,128)
  const long total = 65536L * 128;
  const long stride = (long)gridDim.x * blockDim.x;
  for (long i = (long)blockIdx.x * blockDim.x + threadIdx.x; i < total; i += stride) {
    const long r = i >> 7; const int c = (int)(i & 127);
    const long b = r >> 8; const long d = r & 255;
    out[i] = (_Float16)state[(((b << 1)) * 256 + d) * 128 + c];
  }
}

__global__ void cvt_s2(const float* __restrict__ state, _Float16* __restrict__ out) {
  // s2[b,c] = state[((b*2+1)*256 + 0)*128 + c]; out (256,128)
  const long total = 256L * 128;
  const long stride = (long)gridDim.x * blockDim.x;
  for (long i = (long)blockIdx.x * blockDim.x + threadIdx.x; i < total; i += stride) {
    const long b = i >> 7; const int c = (int)(i & 127);
    out[i] = (_Float16)state[((b * 2 + 1) * 256) * 128 + c];
  }
}

// ---------------------------------------------------------------------------
// Global-scalar BN statistics (deterministic two-stage reduction)
// ---------------------------------------------------------------------------
__global__ void reduce_partial(const float* __restrict__ pre, long rows, int valid,
                               int ld, float* __restrict__ part) {
  __shared__ float ss[256], sq[256];
  const long total = rows * valid;
  const long stride = (long)gridDim.x * blockDim.x;
  float s = 0.f, q = 0.f;
  for (long i = (long)blockIdx.x * blockDim.x + threadIdx.x; i < total; i += stride) {
    const long r = i / valid; const int c = (int)(i - r * (long)valid);
    const float v = pre[r * (long)ld + c];
    s += v; q += v * v;
  }
  ss[threadIdx.x] = s; sq[threadIdx.x] = q;
  __syncthreads();
  for (int o = 128; o > 0; o >>= 1) {
    if (threadIdx.x < o) { ss[threadIdx.x] += ss[threadIdx.x + o]; sq[threadIdx.x] += sq[threadIdx.x + o]; }
    __syncthreads();
  }
  if (threadIdx.x == 0) { part[2 * blockIdx.x] = ss[0]; part[2 * blockIdx.x + 1] = sq[0]; }
}

__global__ void reduce_final(const float* __restrict__ part, int nb, float inv_count,
                             float* __restrict__ stats) {
  __shared__ float ss[256], sq[256];
  float s = 0.f, q = 0.f;
  for (int i = threadIdx.x; i < nb; i += 256) { s += part[2 * i]; q += part[2 * i + 1]; }
  ss[threadIdx.x] = s; sq[threadIdx.x] = q;
  __syncthreads();
  for (int o = 128; o > 0; o >>= 1) {
    if (threadIdx.x < o) { ss[threadIdx.x] += ss[threadIdx.x + o]; sq[threadIdx.x] += sq[threadIdx.x + o]; }
    __syncthreads();
  }
  if (threadIdx.x == 0) {
    const float m = ss[0] * inv_count;
    const float v = sq[0] * inv_count - m * m;
    stats[0] = m;
    stats[1] = rsqrtf(v + 1e-5f);
  }
}

// ---------------------------------------------------------------------------
// BN + tanh apply variants
// ---------------------------------------------------------------------------
__global__ void bn_tanh_h16(const float* __restrict__ pre, long rows, int valid, int ldin,
                            _Float16* __restrict__ out, int ldout,
                            const float* __restrict__ stats,
                            const float* __restrict__ gam, const float* __restrict__ bet, int bi) {
  const float m = stats[0], rs = stats[1], g = gam[bi], b = bet[bi];
  const long total = rows * ldout;
  const long stride = (long)gridDim.x * blockDim.x;
  for (long i = (long)blockIdx.x * blockDim.x + threadIdx.x; i < total; i += stride) {
    const long r = i / ldout; const int c = (int)(i - r * (long)ldout);
    float v = 0.f;
    if (c < valid) v = tanhf(g * (pre[r * (long)ldin + c] - m) * rs + b);
    out[i] = (_Float16)v;
  }
}

// L2 output: apply BN+tanh and the swapaxes(1,2) transpose (per-batch 256x256)
__global__ void bn_tanh_h16_tr(const float* __restrict__ pre, _Float16* __restrict__ out,
                               const float* __restrict__ stats,
                               const float* __restrict__ gam, const float* __restrict__ bet, int bi) {
  const float m = stats[0], rs = stats[1], g = gam[bi], b = bet[bi];
  const long total = 65536L * 256;
  const long stride = (long)gridDim.x * blockDim.x;
  for (long i = (long)blockIdx.x * blockDim.x + threadIdx.x; i < total; i += stride) {
    const long r = i >> 8; const int c = (int)(i & 255);
    const long bb = r >> 8; const int d = (int)(r & 255);
    const float v = tanhf(g * (pre[i] - m) * rs + b);
    out[((bb << 8) + c) * 256L + d] = (_Float16)v;
  }
}

__global__ void bn_tanh_f32v(const float* __restrict__ pre, long n, float* __restrict__ out,
                             const float* __restrict__ stats,
                             const float* __restrict__ gam, const float* __restrict__ bet, int bi) {
  const float m = stats[0], rs = stats[1], g = gam[bi], b = bet[bi];
  const long stride = (long)gridDim.x * blockDim.x;
  for (long i = (long)blockIdx.x * blockDim.x + threadIdx.x; i < n; i += stride)
    out[i] = tanhf(g * (pre[i] - m) * rs + b);
}

// y-path merge: x = x + tanh(bn(y2)); keep f32 + f16 copies of x
__global__ void bn_tanh_add(const float* __restrict__ pre, long n,
                            float* __restrict__ xf, _Float16* __restrict__ xh,
                            const float* __restrict__ stats,
                            const float* __restrict__ gam, const float* __restrict__ bet, int bi) {
  const float m = stats[0], rs = stats[1], g = gam[bi], b = bet[bi];
  const long stride = (long)gridDim.x * blockDim.x;
  for (long i = (long)blockIdx.x * blockDim.x + threadIdx.x; i < n; i += stride) {
    const float v = tanhf(g * (pre[i] - m) * rs + b);
    const float s = xf[i] + v;
    xf[i] = s; xh[i] = (_Float16)s;
  }
}

// residual block epilogue: x = tanh(bn(h) + x)
__global__ void bn_res_tanh(const float* __restrict__ pre, long n,
                            float* __restrict__ xf, _Float16* __restrict__ xh,
                            const float* __restrict__ stats,
                            const float* __restrict__ gam, const float* __restrict__ bet, int bi) {
  const float m = stats[0], rs = stats[1], g = gam[bi], b = bet[bi];
  const long stride = (long)gridDim.x * blockDim.x;
  for (long i = (long)blockIdx.x * blockDim.x + threadIdx.x; i < n; i += stride) {
    const float v = tanhf(g * (pre[i] - m) * rs + b + xf[i]);
    xf[i] = v; xh[i] = (_Float16)v;
  }
}

// L8: K=5 -> plain VALU dot (too small for WMMA)
__global__ void lin8_k(const _Float16* __restrict__ act, int ld,
                       const float* __restrict__ W8, const float* __restrict__ b8,
                       float* __restrict__ out, long n) {
  const long i = (long)blockIdx.x * blockDim.x + threadIdx.x;
  if (i < n) {
    float s = b8[0];
#pragma unroll
    for (int k = 0; k < 5; ++k) s += (float)act[i * (long)ld + k] * W8[k];
    out[i] = s;
  }
}

__global__ void logsoftmax_k(const float* __restrict__ pre, int ld, int n,
                             float* __restrict__ out) {
  const int row = blockIdx.x;
  const float* __restrict__ p = pre + (long)row * ld;
  __shared__ float red[256];
  const int tid = threadIdx.x;
  float mx = -3.4e38f;
  for (int c = tid; c < n; c += 256) mx = fmaxf(mx, p[c]);
  red[tid] = mx; __syncthreads();
  for (int o = 128; o > 0; o >>= 1) { if (tid < o) red[tid] = fmaxf(red[tid], red[tid + o]); __syncthreads(); }
  mx = red[0]; __syncthreads();
  float s = 0.f;
  for (int c = tid; c < n; c += 256) s += expf(p[c] - mx);
  red[tid] = s; __syncthreads();
  for (int o = 128; o > 0; o >>= 1) { if (tid < o) red[tid] += red[tid + o]; __syncthreads(); }
  const float lse = logf(red[0]) + mx;
  for (int c = tid; c < n; c += 256) out[(long)row * n + c] = p[c] - lse;
}

__global__ void val_k(const float* __restrict__ xf, const float* __restrict__ Wv,
                      const float* __restrict__ bv, float* __restrict__ out) {
  const int m = threadIdx.x;   // 256 rows
  float s = bv[0];
  for (int k = 0; k < 256; ++k) s += xf[(long)m * 256 + k] * Wv[k];
  out[m] = 0.5f * tanhf(s) + 0.5f;
}

// ---------------------------------------------------------------------------
// Host-side pipeline
// ---------------------------------------------------------------------------
static void gemm(hipStream_t st, const _Float16* A, int lda, const _Float16* W, int ldw,
                 const float* bias, float* C, int ldc, long M, int N, int Kpad) {
  const int nbn = (N + 63) / 64;                 // weight buffers padded to 64 rows
  const int blocks = (int)((M / 128) * nbn);
  const size_t shmem = (size_t)64 * Kpad * sizeof(_Float16);
  wmma_gemm_lds<<<blocks, 256, shmem, st>>>(A, lda, W, ldw, bias, C, ldc, nbn, M, N, Kpad);
}

extern "C" void kernel_launch(void* const* d_in, const int* in_sizes, int n_in,
                              void* d_out, int out_size, void* d_ws, size_t ws_size,
                              hipStream_t stream) {
  const float* W1  = (const float*)d_in[0];   const float* b1  = (const float*)d_in[1];
  const float* W2  = (const float*)d_in[2];   const float* b2  = (const float*)d_in[3];
  const float* W3  = (const float*)d_in[4];   const float* b3  = (const float*)d_in[5];
  const float* W4  = (const float*)d_in[6];   const float* b4  = (const float*)d_in[7];
  const float* W5  = (const float*)d_in[8];   const float* b5  = (const float*)d_in[9];
  const float* W6  = (const float*)d_in[10];  const float* b6  = (const float*)d_in[11];
  const float* W7  = (const float*)d_in[12];  const float* b7  = (const float*)d_in[13];
  const float* W8  = (const float*)d_in[14];  const float* b8  = (const float*)d_in[15];
  const float* We1 = (const float*)d_in[16];  const float* be1 = (const float*)d_in[17];
  const float* We2 = (const float*)d_in[18];  const float* be2 = (const float*)d_in[19];
  const float* resW = (const float*)d_in[20]; const float* resb = (const float*)d_in[21];
  const float* Wa  = (const float*)d_in[22];  const float* ba  = (const float*)d_in[23];
  const float* Wv  = (const float*)d_in[24];  const float* bv  = (const float*)d_in[25];
  const float* gam = (const float*)d_in[26];  const float* bet = (const float*)d_in[27];
  const float* state = (const float*)d_in[28];
  float* out = (float*)d_out;

  // bump allocator over d_ws
  uintptr_t cur = (uintptr_t)d_ws;
  auto alloc = [&](size_t bytes) -> void* {
    uintptr_t r = (cur + 255) & ~(uintptr_t)255;
    cur = r + bytes;
    return (void*)r;
  };

  _Float16* s1h  = (_Float16*)alloc(65536L * 128 * 2);
  _Float16* actA = (_Float16*)alloc(65536L * 352 * 2);
  _Float16* actB = (_Float16*)alloc(65536L * 352 * 2);
  float*    pre  = (float*)   alloc(65536L * 352 * 4);   // reused per layer (also head 256x8144)
  float*    part = (float*)   alloc(2L * RED_NB * 4);
  float*    stats= (float*)   alloc(2 * 4);
  float*    xf   = (float*)   alloc(65536L * 4);
  _Float16* xh   = (_Float16*)alloc(65536L * 2);
  _Float16* s2h  = (_Float16*)alloc(256L * 128 * 2);
  _Float16* yact = (_Float16*)alloc(256L * 192 * 2);
  // f16 weights, rows padded to multiples of 64 (zero-filled)
  _Float16* W1h  = (_Float16*)alloc(192L * 128 * 2);
  _Float16* W2h  = (_Float16*)alloc(256L * 192 * 2);
  _Float16* W3h  = (_Float16*)alloc(384L * 256 * 2);
  _Float16* W4h  = (_Float16*)alloc(256L * 352 * 2);
  _Float16* W5h  = (_Float16*)alloc(128L * 256 * 2);
  _Float16* W6h  = (_Float16*)alloc(64L  * 128 * 2);
  _Float16* W7h  = (_Float16*)alloc(64L  * 64  * 2);
  _Float16* We1h = (_Float16*)alloc(192L * 128 * 2);
  _Float16* We2h = (_Float16*)alloc(256L * 192 * 2);
  _Float16* resWh= (_Float16*)alloc(5L * 256 * 256 * 2);
  _Float16* Wah  = (_Float16*)alloc(8192L * 256 * 2);

  auto bnstats = [&](const float* p, long rows, int valid, int ld) {
    reduce_partial<<<RED_NB, 256, 0, stream>>>(p, rows, valid, ld, part);
    reduce_final<<<1, 256, 0, stream>>>(part, RED_NB, (float)(1.0 / ((double)rows * valid)), stats);
  };

  // --- weight conversion (f32 -> padded f16) ---
  cvt_w<<<EW_GRID, EW_BLK, 0, stream>>>(W1, 170, 128, W1h, 192, 128);
  cvt_w<<<EW_GRID, EW_BLK, 0, stream>>>(W2, 256, 170, W2h, 256, 192);
  cvt_w<<<EW_GRID, EW_BLK, 0, stream>>>(W3, 341, 256, W3h, 384, 256);
  cvt_w<<<EW_GRID, EW_BLK, 0, stream>>>(W4, 256, 341, W4h, 256, 352);
  cvt_w<<<EW_GRID, EW_BLK, 0, stream>>>(W5, 128, 256, W5h, 128, 256);
  cvt_w<<<EW_GRID, EW_BLK, 0, stream>>>(W6, 64, 128, W6h, 64, 128);
  cvt_w<<<EW_GRID, EW_BLK, 0, stream>>>(W7, 5, 64, W7h, 64, 64);
  cvt_w<<<EW_GRID, EW_BLK, 0, stream>>>(We1, 170, 128, We1h, 192, 128);
  cvt_w<<<EW_GRID, EW_BLK, 0, stream>>>(We2, 256, 170, We2h, 256, 192);
  for (int i = 0; i < 5; ++i)
    cvt_w<<<EW_GRID, EW_BLK, 0, stream>>>(resW + (long)i * 65536, 256, 256,
                                          resWh + (long)i * 65536, 256, 256);
  cvt_w<<<EW_GRID, EW_BLK, 0, stream>>>(Wa, 8129, 256, Wah, 8192, 256);

  // --- x path (M = 65536) ---
  cvt_s1<<<EW_GRID, EW_BLK, 0, stream>>>(state, s1h);
  gemm(stream, s1h, 128, W1h, 128, b1, pre, 176, 65536, 170, 128);
  bnstats(pre, 65536, 170, 176);
  bn_tanh_h16<<<EW_GRID, EW_BLK, 0, stream>>>(pre, 65536, 170, 176, actA, 192, stats, gam, bet, 0);

  gemm(stream, actA, 192, W2h, 192, b2, pre, 256, 65536, 256, 192);
  bnstats(pre, 65536, 256, 256);
  bn_tanh_h16_tr<<<EW_GRID, EW_BLK, 0, stream>>>(pre, actB, stats, gam, bet, 1);  // + swapaxes

  gemm(stream, actB, 256, W3h, 256, b3, pre, 352, 65536, 341, 256);
  bnstats(pre, 65536, 341, 352);
  bn_tanh_h16<<<EW_GRID, EW_BLK, 0, stream>>>(pre, 65536, 341, 352, actA, 352, stats, gam, bet, 2);

  gemm(stream, actA, 352, W4h, 352, b4, pre, 256, 65536, 256, 352);
  bnstats(pre, 65536, 256, 256);
  bn_tanh_h16<<<EW_GRID, EW_BLK, 0, stream>>>(pre, 65536, 256, 256, actB, 256, stats, gam, bet, 3);

  gemm(stream, actB, 256, W5h, 256, b5, pre, 128, 65536, 128, 256);
  bnstats(pre, 65536, 128, 128);
  bn_tanh_h16<<<EW_GRID, EW_BLK, 0, stream>>>(pre, 65536, 128, 128, actA, 128, stats, gam, bet, 4);

  gemm(stream, actA, 128, W6h, 128, b6, pre, 64, 65536, 64, 128);
  bnstats(pre, 65536, 64, 64);
  bn_tanh_h16<<<EW_GRID, EW_BLK, 0, stream>>>(pre, 65536, 64, 64, actB, 64, stats, gam, bet, 5);

  gemm(stream, actB, 64, W7h, 64, b7, pre, 16, 65536, 5, 64);
  bnstats(pre, 65536, 5, 16);
  bn_tanh_h16<<<EW_GRID, EW_BLK, 0, stream>>>(pre, 65536, 5, 16, actA, 8, stats, gam, bet, 6);

  lin8_k<<<256, 256, 0, stream>>>(actA, 8, W8, b8, pre, 65536);
  bnstats(pre, 65536, 1, 1);
  bn_tanh_f32v<<<EW_GRID, EW_BLK, 0, stream>>>(pre, 65536, xf, stats, gam, bet, 7);

  // --- y path (M = 256) ---
  cvt_s2<<<128, EW_BLK, 0, stream>>>(state, s2h);
  gemm(stream, s2h, 128, We1h, 128, be1, pre, 176, 256, 170, 128);
  bnstats(pre, 256, 170, 176);
  bn_tanh_h16<<<EW_GRID, EW_BLK, 0, stream>>>(pre, 256, 170, 176, yact, 192, stats, gam, bet, 8);

  gemm(stream, yact, 192, We2h, 192, be2, pre, 256, 256, 256, 192);
  bnstats(pre, 256, 256, 256);
  bn_tanh_add<<<EW_GRID, EW_BLK, 0, stream>>>(pre, 65536, xf, xh, stats, gam, bet, 9);

  // --- residual tower ---
  for (int i = 0; i < 5; ++i) {
    gemm(stream, xh, 256, resWh + (long)i * 65536, 256, resb + (long)i * 256,
         pre, 256, 256, 256, 256);
    bnstats(pre, 256, 256, 256);
    bn_res_tanh<<<EW_GRID, EW_BLK, 0, stream>>>(pre, 65536, xf, xh, stats, gam, bet, 10);
  }

  // --- policy head: (256x256)x(256->8129) + log_softmax ---
  gemm(stream, xh, 256, Wah, 256, ba, pre, 8144, 256, 8129, 256);
  logsoftmax_k<<<256, 256, 0, stream>>>(pre, 8144, 8129, out);

  // --- value head ---
  val_k<<<1, 256, 0, stream>>>(xf, Wv, bv, out + 256L * 8129);
}
